// Decoder_84636625535251
// MI455X (gfx1250) — matrix operand order repaired and checked by reading, not compiled, not run on hardware
//
#include <hip/hip_runtime.h>

typedef __attribute__((ext_vector_type(16))) __bf16 v16bf;
typedef __attribute__((ext_vector_type(8)))  __bf16 v8bf;
typedef __attribute__((ext_vector_type(8)))  float  v8f;
typedef __attribute__((ext_vector_type(4)))  float  v4f;

#define BSZ 256
#define TT  512
#define INS 1024
#define HH  64
#define BT  (BSZ * TT)

static __device__ __forceinline__ v8f wmma_bf16(v16bf a, v16bf b, v8f c) {
  // (neg_a, A, neg_b, B, c_mod, C, reuse_a, reuse_b)
  return __builtin_amdgcn_wmma_f32_16x16x32_bf16(false, a, false, b, (short)0, c,
                                                 false, false);
}

// 16x32 A-tile (MxK) from row-major fp32, converting to bf16.
// A layout used here (same k order as B): lane l holds row m=l&15; element j -> K = 16*(l>>4) + j (contiguous)
static __device__ __forceinline__ v16bf load_a_f32(const float* src, int stride, int lane) {
  const int m  = lane & 15;
  const int ko = (lane >> 4) << 4;  // 0 or 16
  const float* p = src + (long)m * stride + ko;
  v4f a0 = *(const v4f*)(p + 0);
  v4f a1 = *(const v4f*)(p + 4);
  v4f b0 = *(const v4f*)(p + 8);
  v4f b1 = *(const v4f*)(p + 12);
  v16bf r;
#pragma unroll
  for (int i = 0; i < 4; ++i) {
    r[i]      = (__bf16)a0[i];
    r[i + 4]  = (__bf16)a1[i];
    r[i + 8]  = (__bf16)b0[i];
    r[i + 12] = (__bf16)b1[i];
  }
  return r;
}

// 16x32 A-tile from row-major bf16.
static __device__ __forceinline__ v16bf load_a_bf16(const __bf16* src, int stride, int lane) {
  const int m  = lane & 15;
  const int ko = (lane >> 4) << 4;
  const __bf16* p = src + (long)m * stride + ko;
  v8bf lo = *(const v8bf*)(p);
  v8bf hi = *(const v8bf*)(p + 8);
  v16bf r;
#pragma unroll
  for (int i = 0; i < 8; ++i) { r[i] = lo[i]; r[i + 8] = hi[i]; }
  return r;
}

// 32x16 B-tile (KxN) where column n of B = row n of W (row-major [N x K]).
// B layout used here (same k order as A): lane l holds col n=l&15; element j -> K = 16*(l>>4) + j (contiguous)
static __device__ __forceinline__ v16bf load_b_bf16(const __bf16* w, int stride, int lane) {
  const int n  = lane & 15;
  const int ko = (lane >> 4) << 4;  // 0 or 16
  const __bf16* p = w + (long)n * stride + ko;
  v8bf lo = *(const v8bf*)(p);
  v8bf hi = *(const v8bf*)(p + 8);
  v16bf r;
#pragma unroll
  for (int i = 0; i < 8; ++i) { r[i] = lo[i]; r[i + 8] = hi[i]; }
  return r;
}

// ---- fp32 -> bf16 weight conversion (tiny) ----
__global__ void k_cvt(const float* __restrict__ src, __bf16* __restrict__ dst, int n) {
  int i = blockIdx.x * blockDim.x + threadIdx.x;
  if (i < n) dst[i] = (__bf16)src[i];
}

// ---- Stage 1: ext[bt, h] = x[bt, :] @ W_in[h, :] + b_in[h] ----
// one wave = 32 rows x 64 cols (2 m-tiles x 4 n-tiles); all loads for a k-step are
// issued before any WMMA so loadcnt waits grade down instead of hitting 0 each time.
// HBM-bound on x (512 MB fp32 -> ~22 us floor at 23.3 TB/s).
__global__ __launch_bounds__(256) void k_in_gemm(const float* __restrict__ x,
                                                 const __bf16* __restrict__ Winb,
                                                 const float* __restrict__ b_in,
                                                 float* __restrict__ ext) {
  const int lane = threadIdx.x & 31;
  const int wave = threadIdx.x >> 5;
  const long row0 = ((long)blockIdx.x * 8 + wave) * 32;
  const float* xs0 = x + row0 * INS;
  const float* xs1 = x + (row0 + 16) * INS;

  v8f acc[2][4] = {};
  for (int k = 0; k < INS; k += 32) {
    // Issue every load for this k-step first (A: 8x b128 fp32, B: 8x b128 bf16).
    v16bf a0 = load_a_f32(xs0 + k, INS, lane);
    v16bf a1 = load_a_f32(xs1 + k, INS, lane);
    v16bf b0 = load_b_bf16(Winb + 0 * 16 * INS + k, INS, lane);
    v16bf b1 = load_b_bf16(Winb + 1 * 16 * INS + k, INS, lane);
    v16bf b2 = load_b_bf16(Winb + 2 * 16 * INS + k, INS, lane);
    v16bf b3 = load_b_bf16(Winb + 3 * 16 * INS + k, INS, lane);
    acc[0][0] = wmma_bf16(a0, b0, acc[0][0]);
    acc[1][0] = wmma_bf16(a1, b0, acc[1][0]);
    acc[0][1] = wmma_bf16(a0, b1, acc[0][1]);
    acc[1][1] = wmma_bf16(a1, b1, acc[1][1]);
    acc[0][2] = wmma_bf16(a0, b2, acc[0][2]);
    acc[1][2] = wmma_bf16(a1, b2, acc[1][2]);
    acc[0][3] = wmma_bf16(a0, b3, acc[0][3]);
    acc[1][3] = wmma_bf16(a1, b3, acc[1][3]);
  }
  // C/D layout: lane holds col n=l&15, rows (l>>4)*8 + r
  const int n  = lane & 15;
  const int rs = (lane >> 4) << 3;
#pragma unroll
  for (int mt = 0; mt < 2; ++mt) {
#pragma unroll
    for (int nt = 0; nt < 4; ++nt) {
      float bias = b_in[nt * 16 + n];
#pragma unroll
      for (int r = 0; r < 8; ++r)
        ext[(row0 + mt * 16 + rs + r) * HH + nt * 16 + n] = acc[mt][nt][r] + bias;
    }
  }
}

// ---- Stage 2: sequential scan, transposed form G = state^T (64 x 16batch) ----
// G_t = relu(W_rec @ (G_{t-1} + E_t^T) + b_rec). D->B relayout per step needs only a
// half-wave shfl_xor(16) + selects: register-resident recurrence, no LDS.
__global__ __launch_bounds__(32) void k_recur(const float* __restrict__ ext,
                                              const __bf16* __restrict__ Wrb,
                                              const float* __restrict__ b_rec,
                                              __bf16* __restrict__ states) {
  const int lane = threadIdx.x & 31;
  const int b0 = blockIdx.x * 16;       // 16 batch rows per wave
  const int hsel = lane >> 4;           // half-wave select
  const int nl = lane & 15;             // column (batch) within tile
  const int rs = hsel << 3;
  const bool upper = (hsel != 0);

  // Preload A = W_rec as 4(mt: h_out) x 2(kt: h_in) tiles, registers only.
  v16bf awr[4][2];
#pragma unroll
  for (int mt = 0; mt < 4; ++mt)
#pragma unroll
    for (int kt = 0; kt < 2; ++kt)
      awr[mt][kt] = load_a_bf16(Wrb + mt * 16 * HH + kt * 32, HH, lane);

  // Bias as C-init: c[r] = b_rec[row m = mt*16 + rs + r]
  v8f cb[4];
#pragma unroll
  for (int mt = 0; mt < 4; ++mt)
#pragma unroll
    for (int r = 0; r < 8; ++r) cb[mt][r] = b_rec[mt * 16 + rs + r];

  v8f g[4];
#pragma unroll
  for (int mt = 0; mt < 4; ++mt) g[mt] = {};

  const float* eb = ext + (long)(b0 + nl) * TT * HH;  // ext rows for my column
  for (int t = 0; t < TT; ++t) {
    const float* ep = eb + (long)t * HH;
    if (t + 1 < TT) __builtin_prefetch(ep + HH, 0, 0);

    // Build B tiles (K=h_in, N=batch) from previous D (g) + ext, fp32 add then cvt.
    v16bf btile[2];
#pragma unroll
    for (int kt = 0; kt < 2; ++kt) {
      const int kbase = kt * 32 + (hsel << 4);
#pragma unroll
      for (int j = 0; j < 8; ++j) {
        float send = upper ? g[2 * kt][j] : g[2 * kt + 1][j];
        float recv = __shfl_xor(send, 16, 32);
        float lo_s = upper ? recv : g[2 * kt][j];          // K = kbase + j
        float hi_s = upper ? g[2 * kt + 1][j] : recv;      // K = kbase + 8 + j
        btile[kt][j]     = (__bf16)(lo_s + ep[kbase + j]);
        btile[kt][j + 8] = (__bf16)(hi_s + ep[kbase + 8 + j]);
      }
    }

#pragma unroll
    for (int mt = 0; mt < 4; ++mt) {
      v8f c = cb[mt];
      c = wmma_bf16(awr[mt][0], btile[0], c);
      c = wmma_bf16(awr[mt][1], btile[1], c);
#pragma unroll
      for (int r = 0; r < 8; ++r) c[r] = fmaxf(c[r], 0.0f);
      g[mt] = c;
    }

    // states[b0+nl, t, :]: element (m = mt*16+rs+r, n = nl); 8 bf16 contiguous -> 16B store
    __bf16* sp = states + ((long)(b0 + nl) * TT + t) * HH;
#pragma unroll
    for (int mt = 0; mt < 4; ++mt) {
      v8bf sv;
#pragma unroll
      for (int r = 0; r < 8; ++r) sv[r] = (__bf16)g[mt][r];
      *(v8bf*)(sp + mt * 16 + rs) = sv;
    }
  }
}

// ---- Stage 3: h = relu(states @ W_o1^T + b_o1); out = h @ W_o2^T + b_o2 ----
__global__ __launch_bounds__(256) void k_head(const __bf16* __restrict__ states,
                                              const __bf16* __restrict__ Wo1b,
                                              const float* __restrict__ b_o1,
                                              const float* __restrict__ W_o2,
                                              const float* __restrict__ b_o2,
                                              float* __restrict__ out) {
  const int lane = threadIdx.x & 31;
  const int wave = threadIdx.x >> 5;
  const long row0 = ((long)blockIdx.x * 8 + wave) * 16;
  const __bf16* s = states + row0 * HH;

  v16bf a0 = load_a_bf16(s, HH, lane);
  v16bf a1 = load_a_bf16(s + 32, HH, lane);

  const int n = lane & 15;
  v8f hacc[2];
#pragma unroll
  for (int nt = 0; nt < 2; ++nt) {
    v8f c = {};
    float bias = b_o1[nt * 16 + n];
#pragma unroll
    for (int r = 0; r < 8; ++r) c[r] = bias;
    c = wmma_bf16(a0, load_b_bf16(Wo1b + nt * 16 * HH, HH, lane), c);
    c = wmma_bf16(a1, load_b_bf16(Wo1b + nt * 16 * HH + 32, HH, lane), c);
#pragma unroll
    for (int r = 0; r < 8; ++r) c[r] = fmaxf(c[r], 0.0f);
    hacc[nt] = c;
  }

  // out[m, cc] = sum_o h[m, o] * W_o2[cc, o] + b_o2[cc]; reduce 16 lanes (o dim)
  const int rs = (lane >> 4) << 3;
#pragma unroll
  for (int cc = 0; cc < 2; ++cc) {
    float w0 = W_o2[cc * 32 + n];
    float w1 = W_o2[cc * 32 + 16 + n];
    float bias = b_o2[cc];
#pragma unroll
    for (int r = 0; r < 8; ++r) {
      float p = hacc[0][r] * w0 + hacc[1][r] * w1;
      p += __shfl_xor(p, 1, 32);
      p += __shfl_xor(p, 2, 32);
      p += __shfl_xor(p, 4, 32);
      p += __shfl_xor(p, 8, 32);
      if (n == 0) out[(row0 + rs + r) * 2 + cc] = p + bias;
    }
  }
}

extern "C" void kernel_launch(void* const* d_in, const int* in_sizes, int n_in,
                              void* d_out, int out_size, void* d_ws, size_t ws_size,
                              hipStream_t stream) {
  const float* x     = (const float*)d_in[0];
  const float* W_in  = (const float*)d_in[1];
  const float* b_in  = (const float*)d_in[2];
  const float* W_rec = (const float*)d_in[3];
  const float* b_rec = (const float*)d_in[4];
  const float* W_o1  = (const float*)d_in[5];
  const float* b_o1  = (const float*)d_in[6];
  const float* W_o2  = (const float*)d_in[7];
  const float* b_o2  = (const float*)d_in[8];
  float* outp = (float*)d_out;

  char* ws = (char*)d_ws;
  float*  extf    = (float*)(ws);                               // BT*H fp32  (32 MB)
  __bf16* statesb = (__bf16*)(ws + (size_t)BT * HH * 4);        // BT*H bf16  (16 MB)
  size_t  woff    = (size_t)BT * HH * 6;
  __bf16* Winb    = (__bf16*)(ws + woff);                       // 64*1024 bf16
  __bf16* Wrecb   = (__bf16*)(ws + woff + (size_t)HH * INS * 2);
  __bf16* Wo1b    = (__bf16*)(ws + woff + (size_t)HH * INS * 2 + (size_t)HH * HH * 2);

  k_cvt<<<(HH * INS + 255) / 256, 256, 0, stream>>>(W_in, Winb, HH * INS);
  k_cvt<<<(HH * HH + 255) / 256, 256, 0, stream>>>(W_rec, Wrecb, HH * HH);
  k_cvt<<<((HH / 2) * HH + 255) / 256, 256, 0, stream>>>(W_o1, Wo1b, (HH / 2) * HH);

  k_in_gemm<<<dim3(BT / 256), dim3(256), 0, stream>>>(x, Winb, b_in, extf);
  k_recur<<<dim3(BSZ / 16), dim3(32), 0, stream>>>(extf, Wrecb, b_rec, statesb);
  k_head<<<dim3(BT / 128), dim3(256), 0, stream>>>(statesb, Wo1b, b_o1, W_o2, b_o2, outp);
}
